// ProteinGNN_11871289606631
// MI455X (gfx1250) — matrix operand order repaired and checked by reading, not compiled, not run on hardware
//
#include <hip/hip_runtime.h>
#include <cmath>

typedef __attribute__((ext_vector_type(16))) _Float16 v16h;
typedef __attribute__((ext_vector_type(4)))  _Float16 v4h;
typedef __attribute__((ext_vector_type(8)))  float    v8f;

// ---------------------------------------------------------------------------
// order-preserving float <-> uint encoding for atomicMax on signed floats
// ---------------------------------------------------------------------------
__device__ __forceinline__ unsigned enc_f(float x) {
    unsigned u = __float_as_uint(x);
    return (u & 0x80000000u) ? ~u : (u | 0x80000000u);
}
__device__ __forceinline__ float dec_f(unsigned u) {
    return (u & 0x80000000u) ? __uint_as_float(u & 0x7FFFFFFFu)
                             : __uint_as_float(~u);
}

// ---------------------------------------------------------------------------
// Pack W[K,Kout] (f32 row-major) into WMMA B-fragment order:
//   P[ct][ks][lane][i] = (f16) W[(ks*32 + (lane>>4)*16 + i)*Kout + ct*16 + (lane&15)]
// kss = log2(K/32). One thread per (ct, ks, lane); writes 16 halves (32 B).
// ---------------------------------------------------------------------------
__global__ void pack_w_f16(const float* __restrict__ W, _Float16* __restrict__ P,
                           int kss, int Kout, int total)
{
    int idx = blockIdx.x * blockDim.x + threadIdx.x;
    if (idx >= total) return;
    int lane = idx & 31;
    int rest = idx >> 5;
    int ks   = rest & ((1 << kss) - 1);
    int ct   = rest >> kss;
    int half = lane >> 4, nco = lane & 15;
    const float* Wp = W + (size_t)((ks << 5) + (half << 4)) * Kout + (ct << 4) + nco;
    _Float16* Pp = P + (size_t)idx * 16;
    #pragma unroll
    for (int i = 0; i < 16; ++i) Pp[i] = (_Float16)Wp[(size_t)i * Kout];
}

// ---------------------------------------------------------------------------
// WMMA GEMM: C[n, Kout] = A[n, K] @ W + bias (W pre-packed to fragments),
// optional exact GELU. One 16-row tile per block; wave w owns col tile w.
// blockDim.x = (Kout/16)*32; N % 16 == 0; K % 128 == 0.
// ---------------------------------------------------------------------------
__global__ __launch_bounds__(512)
void wmma_gemm_bias_act(const float* __restrict__ A, const _Float16* __restrict__ BP,
                        const float* __restrict__ bias, float* __restrict__ C,
                        int K, int Kout, int act /*0=none,1=gelu*/)
{
    __shared__ _Float16 sA[16][128];          // 16 rows x 128 K-columns, f16
    const int rowtile = blockIdx.x;
    const int tid  = threadIdx.x;
    const int wave = tid >> 5;
    const int lane = tid & 31;
    const int half = lane >> 4;               // 16-lane group
    const int mrow = lane & 15;               // A row within tile
    const int ct   = wave;                    // column tile
    const int nco  = lane & 15;               // output column within tile
    const int ksteps = K >> 5;

    v8f acc = {0.f, 0.f, 0.f, 0.f, 0.f, 0.f, 0.f, 0.f};
    const float* Arow  = A + (size_t)rowtile * 16 * K;
    const v16h*  Bfrag = (const v16h*)BP + (size_t)ct * ksteps * 32 + lane;

    // vectorized staging: each thread moves a float4 -> 4 halves (ds_store_b64)
    const int rstep = blockDim.x >> 5;        // rows per pass (8 or 16)
    const int r0 = tid >> 5;
    const int c4 = lane << 2;                 // float4-aligned column
    const size_t pstride = (size_t)rstep * K;

    for (int k0 = 0; k0 < K; k0 += 128) {
        const float* p = Arow + (size_t)r0 * K + k0 + c4;
        _Float16* sp = &sA[r0][c4];
        for (int r = r0; r < 16; r += rstep) {
            float4 fv = *(const float4*)p;
            v4h hv = { (_Float16)fv.x, (_Float16)fv.y, (_Float16)fv.z, (_Float16)fv.w };
            *(v4h*)sp = hv;
            p  += pstride;
            sp += rstep * 128;
        }
        __syncthreads();
        #pragma unroll
        for (int s = 0; s < 4; ++s) {
            // A fragment: contiguous k = 16*half + i (matches packed B) -> two b128 LDS loads
            v16h a;
            #pragma unroll
            for (int i = 0; i < 16; ++i) {
                int kk = half * 16 + i;
                a[i] = sA[mrow][s * 32 + kk];
            }
            // B fragment: one contiguous 32-byte load from packed weights
            v16h b = Bfrag[(size_t)((k0 >> 5) + s) * 32];
            acc = __builtin_amdgcn_wmma_f32_16x16x32_f16(
                      false, a, false, b, (short)0, acc, false, false);
        }
        __syncthreads();
    }

    const int col = ct * 16 + nco;
    const float bb = bias ? bias[col] : 0.0f;
    #pragma unroll
    for (int v = 0; v < 8; ++v) {
        int row = rowtile * 16 + half * 8 + v;   // C layout: VGPR v -> M = v + 8*half
        float x = acc[v] + bb;
        if (act == 1) x = 0.5f * x * (1.0f + erff(x * 0.70710678118654752f));
        C[(size_t)row * Kout + col] = x;
    }
}

// ---------------------------------------------------------------------------
// per-(node, head) attention logits; Hs = log2(H), HDs = log2(HD).
// h index (n*H + hh)*HD == idx*HD, so a single shifted base suffices.
// ---------------------------------------------------------------------------
__global__ void gat_scores(const float* __restrict__ h, const float* __restrict__ a_src,
                           const float* __restrict__ a_dst, float* __restrict__ s_src,
                           float* __restrict__ s_dst, int Hs, int HDs, int total)
{
    int idx = blockIdx.x * blockDim.x + threadIdx.x;   // n*H + hh
    if (idx >= total) return;
    int hh = idx & ((1 << Hs) - 1);
    const int HD = 1 << HDs;
    const float* hp  = h + ((size_t)idx << HDs);
    const float* asp = a_src + (hh << HDs);
    const float* adp = a_dst + (hh << HDs);
    float ss = 0.f, sd = 0.f;
    for (int d = 0; d < HD; ++d) { float v = hp[d]; ss += v * asp[d]; sd += v * adp[d]; }
    s_src[idx] = ss;
    s_dst[idx] = sd;
}

__global__ void fill_u32(unsigned* __restrict__ p, unsigned val, int n)
{
    int i = blockIdx.x * blockDim.x + threadIdx.x;
    if (i < n) p[i] = val;
}

// edge list + implicit self loops: e < E -> (ei[e], ei[E+e]); else (e-E, e-E)
__device__ __forceinline__ void edge_src_dst(const int* ei, int e, int E, int& s, int& d)
{
    if (e < E) { s = ei[e]; d = ei[E + e]; } else { s = e - E; d = e - E; }
}

__global__ void edge_max(const int* __restrict__ ei, const float* __restrict__ s_src,
                         const float* __restrict__ s_dst, unsigned* __restrict__ m,
                         int Hs, int E, int nedges)
{
    int idx = blockIdx.x * blockDim.x + threadIdx.x;
    if (idx >= (nedges << Hs)) return;
    int e = idx >> Hs, hh = idx & ((1 << Hs) - 1);
    int src, dst; edge_src_dst(ei, e, E, src, dst);
    float ev = s_src[(src << Hs) + hh] + s_dst[(dst << Hs) + hh];
    ev = ev > 0.f ? ev : 0.2f * ev;                 // leaky_relu(0.2)
    atomicMax(&m[(dst << Hs) + hh], enc_f(ev));
}

__global__ void edge_sum(const int* __restrict__ ei, const float* __restrict__ s_src,
                         const float* __restrict__ s_dst, const unsigned* __restrict__ m,
                         float* __restrict__ denom, int Hs, int E, int nedges)
{
    int idx = blockIdx.x * blockDim.x + threadIdx.x;
    if (idx >= (nedges << Hs)) return;
    int e = idx >> Hs, hh = idx & ((1 << Hs) - 1);
    int src, dst; edge_src_dst(ei, e, E, src, dst);
    float ev = s_src[(src << Hs) + hh] + s_dst[(dst << Hs) + hh];
    ev = ev > 0.f ? ev : 0.2f * ev;
    atomicAdd(&denom[(dst << Hs) + hh], expf(ev - dec_f(m[(dst << Hs) + hh])));
}

// wave-per-edge: lane owns 4 consecutive features (always one head for HD>=4):
// one float4 gather of h[src], alpha computed once per lane, 4 scatter-adds.
__global__ __launch_bounds__(256)
void edge_scatter(const int* __restrict__ ei, const float* __restrict__ s_src,
                  const float* __restrict__ s_dst, const unsigned* __restrict__ m,
                  const float* __restrict__ denom, const float* __restrict__ h,
                  float* __restrict__ out, int Hs, int HDs, int ldo, int coloff,
                  int E, int nedges)
{
    int e = blockIdx.x * 8 + (threadIdx.x >> 5);
    if (e >= nedges) return;
    int lane = threadIdx.x & 31;
    int src, dst; edge_src_dst(ei, e, E, src, dst);
    const int f0 = lane << 2;                       // F == 128 here
    const int hh = f0 >> HDs;
    const float* hrow = h + ((size_t)src << 7);     // h row stride 128
    __builtin_prefetch(hrow + f0, 0, 0);            // global_prefetch_b8
    float ev = s_src[(src << Hs) + hh] + s_dst[(dst << Hs) + hh];
    ev = ev > 0.f ? ev : 0.2f * ev;
    float alpha = expf(ev - dec_f(m[(dst << Hs) + hh])) / denom[(dst << Hs) + hh];
    const float4 hv = *(const float4*)(hrow + f0);
    float* op = out + (size_t)dst * ldo + coloff + f0;
    atomicAdd(op + 0, hv.x * alpha);
    atomicAdd(op + 1, hv.y * alpha);
    atomicAdd(op + 2, hv.z * alpha);
    atomicAdd(op + 3, hv.w * alpha);
}

// out[n, offout+f] = act(in[n, offin+f] + bias[f]); F == 128 (shift 7)
__global__ void bias_act(float* __restrict__ out, const float* __restrict__ in,
                         const float* __restrict__ bias, int ldin, int offin,
                         int ldout, int offout, int act, int total)
{
    int idx = blockIdx.x * blockDim.x + threadIdx.x;
    if (idx >= total) return;
    int n = idx >> 7, f = idx & 127;
    float x = in[(size_t)n * ldin + offin + f] + bias[f];
    if (act == 1) x = x > 0.f ? x : expm1f(x);      // elu(alpha=1)
    out[(size_t)n * ldout + offout + f] = x;
}

// wave-per-row LayerNorm over 256 features, in place
__global__ __launch_bounds__(256)
void layernorm_inplace(float* __restrict__ f, const float* __restrict__ g,
                       const float* __restrict__ b, int n_nodes)
{
    int row = blockIdx.x * 8 + (threadIdx.x >> 5);
    if (row >= n_nodes) return;
    int lane = threadIdx.x & 31;
    float* fr = f + ((size_t)row << 8);
    float vals[8]; float s = 0.f, s2 = 0.f;
    #pragma unroll
    for (int i = 0; i < 8; ++i) {
        float v = fr[lane + 32 * i];
        vals[i] = v; s += v; s2 += v * v;
    }
    #pragma unroll
    for (int o = 16; o > 0; o >>= 1) {
        s  += __shfl_xor(s,  o, 32);
        s2 += __shfl_xor(s2, o, 32);
    }
    float mu  = s  * (1.0f / 256.0f);
    float var = s2 * (1.0f / 256.0f) - mu * mu;
    float rs  = rsqrtf(var + 1e-5f);
    #pragma unroll
    for (int i = 0; i < 8; ++i) {
        int c = lane + 32 * i;
        fr[c] = (vals[i] - mu) * rs * g[c] + b[c];
    }
}

// ---------------------------------------------------------------------------
extern "C" void kernel_launch(void* const* d_in, const int* in_sizes, int n_in,
                              void* d_out, int out_size, void* d_ws, size_t ws_size,
                              hipStream_t stream)
{
    const int IN = 256, HID = 128, OUT = 256, HEADS = 4;
    const int N  = in_sizes[0] / IN;        // 50000
    const int E  = in_sizes[1] / 2;         // 800000
    const int NE = E + N;                   // edges incl. self loops

    const float* x  = (const float*)d_in[0];
    const int* ei_f = (const int*)d_in[1];
    const int* ei_r = (const int*)d_in[2];
    const float *W_f1 = (const float*)d_in[3],  *as_f1 = (const float*)d_in[4],
                *ad_f1 = (const float*)d_in[5], *b_f1  = (const float*)d_in[6],
                *W_f2 = (const float*)d_in[7],  *as_f2 = (const float*)d_in[8],
                *ad_f2 = (const float*)d_in[9], *b_f2  = (const float*)d_in[10],
                *W_r1 = (const float*)d_in[11], *as_r1 = (const float*)d_in[12],
                *ad_r1 = (const float*)d_in[13],*b_r1  = (const float*)d_in[14],
                *W_r2 = (const float*)d_in[15], *as_r2 = (const float*)d_in[16],
                *ad_r2 = (const float*)d_in[17],*b_r2  = (const float*)d_in[18],
                *W_fu1 = (const float*)d_in[19],*b_fu1 = (const float*)d_in[20],
                *W_fu2 = (const float*)d_in[21],*b_fu2 = (const float*)d_in[22],
                *ln_g  = (const float*)d_in[23],*ln_b  = (const float*)d_in[24];
    float* out = (float*)d_out;

    // workspace layout (floats)
    float*    ws    = (float*)d_ws;
    float*    hfeat = ws;                              // N*128
    float*    h1    = hfeat + (size_t)N * HID;         // N*128
    float*    agg   = h1    + (size_t)N * HID;         // N*128
    float*    ssrc  = agg   + (size_t)N * HID;         // N*4
    float*    sdst  = ssrc  + (size_t)N * HEADS;       // N*4
    unsigned* mbuf  = (unsigned*)(sdst + (size_t)N * HEADS);   // N*4
    float*    denom = (float*)mbuf + (size_t)N * HEADS;        // N*4
    float*    hcat  = denom + (size_t)N * HEADS;       // N*256
    float*    fmid  = hfeat;  // fusion intermediate aliases hfeat+h1 (dead then)

    // packed f16 weight fragments (after hcat; base is 32 B aligned: 2624*N bytes)
    _Float16* pk    = (_Float16*)(hcat + (size_t)N * OUT);
    _Float16* P_f1  = pk;                         // 256x128 -> 32768 halves
    _Float16* P_f2  = P_f1  + 256 * 128;          // 128x128 -> 16384
    _Float16* P_r1  = P_f2  + 128 * 128;
    _Float16* P_r2  = P_r1  + 256 * 128;
    _Float16* P_fu1 = P_r2  + 128 * 128;          // 256x256 -> 65536
    _Float16* P_fu2 = P_fu1 + 256 * 256;

    const int rowtiles = N / 16;
    auto cdiv = [](int a, int b) { return (a + b - 1) / b; };

    // pack weights into WMMA fragment order (tiny; L2-resident thereafter)
    // kss = log2(K/32): K=256 -> 3, K=128 -> 2
    pack_w_f16<<<cdiv(256*128/16, 256), 256, 0, stream>>>(W_f1,  P_f1,  3, HID, 256*128/16);
    pack_w_f16<<<cdiv(128*128/16, 256), 256, 0, stream>>>(W_f2,  P_f2,  2, HID, 128*128/16);
    pack_w_f16<<<cdiv(256*128/16, 256), 256, 0, stream>>>(W_r1,  P_r1,  3, HID, 256*128/16);
    pack_w_f16<<<cdiv(128*128/16, 256), 256, 0, stream>>>(W_r2,  P_r2,  2, HID, 128*128/16);
    pack_w_f16<<<cdiv(256*256/16, 256), 256, 0, stream>>>(W_fu1, P_fu1, 3, OUT, 256*256/16);
    pack_w_f16<<<cdiv(256*256/16, 256), 256, 0, stream>>>(W_fu2, P_fu2, 3, OUT, 256*256/16);

    // zero the concat buffer (layer-2 scatter target)
    fill_u32<<<cdiv(N * OUT, 256), 256, 0, stream>>>((unsigned*)hcat, 0u, N * OUT);

    auto run_branch = [&](const int* ei, const _Float16* P1, const float* as1,
                          const float* ad1, const float* b1, const _Float16* P2,
                          const float* as2, const float* ad2, const float* b2,
                          int coloff) {
        // ---- GAT layer 1 (4 heads x 32): Hs=2, HDs=5 ----
        wmma_gemm_bias_act<<<rowtiles, 256, 0, stream>>>(x, P1, nullptr, hfeat, IN, HID, 0);
        gat_scores<<<cdiv(N * HEADS, 256), 256, 0, stream>>>(hfeat, as1, ad1, ssrc, sdst, 2, 5, N * HEADS);
        fill_u32<<<cdiv(N * HEADS, 256), 256, 0, stream>>>(mbuf, 0u, N * HEADS);
        fill_u32<<<cdiv(N * HEADS, 256), 256, 0, stream>>>((unsigned*)denom, 0u, N * HEADS);
        fill_u32<<<cdiv(N * HID, 256), 256, 0, stream>>>((unsigned*)agg, 0u, N * HID);
        edge_max<<<cdiv(NE * HEADS, 256), 256, 0, stream>>>(ei, ssrc, sdst, mbuf, 2, E, NE);
        edge_sum<<<cdiv(NE * HEADS, 256), 256, 0, stream>>>(ei, ssrc, sdst, mbuf, denom, 2, E, NE);
        edge_scatter<<<cdiv(NE, 8), 256, 0, stream>>>(ei, ssrc, sdst, mbuf, denom,
                                                      hfeat, agg, 2, 5, HID, 0, E, NE);
        bias_act<<<cdiv(N * HID, 256), 256, 0, stream>>>(h1, agg, b1, HID, 0, HID, 0, 1, N * HID);

        // ---- GAT layer 2 (1 head x 128): Hs=0, HDs=7; aggregate into hcat ----
        wmma_gemm_bias_act<<<rowtiles, 256, 0, stream>>>(h1, P2, nullptr, hfeat, HID, HID, 0);
        gat_scores<<<cdiv(N, 256), 256, 0, stream>>>(hfeat, as2, ad2, ssrc, sdst, 0, 7, N);
        fill_u32<<<cdiv(N, 256), 256, 0, stream>>>(mbuf, 0u, N);
        fill_u32<<<cdiv(N, 256), 256, 0, stream>>>((unsigned*)denom, 0u, N);
        edge_max<<<cdiv(NE, 256), 256, 0, stream>>>(ei, ssrc, sdst, mbuf, 0, E, NE);
        edge_sum<<<cdiv(NE, 256), 256, 0, stream>>>(ei, ssrc, sdst, mbuf, denom, 0, E, NE);
        edge_scatter<<<cdiv(NE, 8), 256, 0, stream>>>(ei, ssrc, sdst, mbuf, denom,
                                                      hfeat, hcat, 0, 7, 2 * HID, coloff, E, NE);
        bias_act<<<cdiv(N * HID, 256), 256, 0, stream>>>(hcat, hcat, b2,
                                                         2 * HID, coloff, 2 * HID, coloff, 0, N * HID);
    };

    run_branch(ei_f, P_f1, as_f1, ad_f1, b_f1, P_f2, as_f2, ad_f2, b_f2, 0);
    run_branch(ei_r, P_r1, as_r1, ad_r1, b_r1, P_r2, as_r2, ad_r2, b_r2, HID);

    // ---- fusion MLP + LayerNorm ----
    wmma_gemm_bias_act<<<rowtiles, 512, 0, stream>>>(hcat, P_fu1, b_fu1, fmid, 2 * HID, OUT, 1);
    wmma_gemm_bias_act<<<rowtiles, 512, 0, stream>>>(fmid, P_fu2, b_fu2, out, OUT, OUT, 0);
    layernorm_inplace<<<cdiv(N, 8), 256, 0, stream>>>(out, ln_g, ln_b, N);
}
